// Attention_70093866270731
// MI455X (gfx1250) — matrix operand order repaired and checked by reading, not compiled, not run on hardware
//
#include <hip/hip_runtime.h>

typedef __attribute__((ext_vector_type(16))) _Float16 v16h;
typedef __attribute__((ext_vector_type(8)))  _Float16 v8h;
typedef __attribute__((ext_vector_type(8)))  float    v8f;
typedef __attribute__((ext_vector_type(4)))  float    v4f;

constexpr int Bb  = 2;
constexpr int Ss  = 2048;
constexpr int Dd  = 1024;
constexpr int Hh  = 16;
constexpr int DHh = 64;

// ---------------------------------------------------------------------------
// helpers
// ---------------------------------------------------------------------------
__device__ __forceinline__ v8f zero8() {
  v8f z;
#pragma unroll
  for (int i = 0; i < 8; ++i) z[i] = 0.0f;
  return z;
}

__device__ __forceinline__ v8f wmma_f16(v16h a, v16h b, v8f c) {
  // D = A(16x32 f16) * B(32x16 f16) + C(16x16 f32)
  return __builtin_amdgcn_wmma_f32_16x16x32_f16(
      /*neg_a=*/false, a, /*neg_b=*/false, b,
      /*c_mod=*/(short)0, c, /*reuse_a=*/false, /*reuse_b=*/false);
}

// A-fragment (16x32 f16) from row-major [m][k] buffer, ld in halves (%8==0).
__device__ __forceinline__ v16h load_frag_a(const _Float16* base, int ld, int lane) {
  const int m  = lane & 15;
  const int kh = (lane >> 4) << 3;            // 0 or 8
  const v8h lo = *(const v8h*)(base + (size_t)m * ld + kh);
  const v8h hi = *(const v8h*)(base + (size_t)m * ld + kh + 16);
  v16h r;
#pragma unroll
  for (int i = 0; i < 8; ++i) { r[i] = lo[i]; r[i + 8] = hi[i]; }
  return r;
}

// B-fragment (32x16 f16) from B^T stored row-major as [n][k], ld in halves.
__device__ __forceinline__ v16h load_frag_b(const _Float16* base, int ld, int lane) {
  const int n  = lane & 15;
  const int kb = (lane >> 4) << 3;            // 0 or 8
  const v8h lo = *(const v8h*)(base + (size_t)n * ld + kb);
  const v8h hi = *(const v8h*)(base + (size_t)n * ld + kb + 16);
  v16h r;
#pragma unroll
  for (int i = 0; i < 8; ++i) { r[i] = lo[i]; r[i + 8] = hi[i]; }
  return r;
}

// ---------------------------------------------------------------------------
// Pre-pass 1: f32 -> f16 bulk convert (for x)
// ---------------------------------------------------------------------------
__global__ __launch_bounds__(256) void convert_f16_kernel(
    const float* __restrict__ src, _Float16* __restrict__ dst) {
  const size_t i = ((size_t)blockIdx.x * 256 + threadIdx.x) * 8;
  const v4f a = *(const v4f*)(src + i);
  const v4f b = *(const v4f*)(src + i + 4);
  v8h o;
#pragma unroll
  for (int j = 0; j < 4; ++j) { o[j] = (_Float16)a[j]; o[j + 4] = (_Float16)b[j]; }
  *(v8h*)(dst + i) = o;
}

// ---------------------------------------------------------------------------
// Pre-pass 2: W[K][N] f32 -> Wt[N][K] f16 tiled transpose (one-time)
// ---------------------------------------------------------------------------
__global__ __launch_bounds__(256) void transpose_f16_kernel(
    const float* __restrict__ src, _Float16* __restrict__ dst, int K, int N) {
  __shared__ float tile[32][33];
  const int tid = threadIdx.x;
  const int nt = blockIdx.x * 32, kt = blockIdx.y * 32;
  const int c = tid & 31, r0 = tid >> 5;          // c: 0..31, r0: 0..7
#pragma unroll
  for (int i = 0; i < 4; ++i)
    tile[r0 + 8 * i][c] = src[(size_t)(kt + r0 + 8 * i) * N + nt + c];
  __syncthreads();
#pragma unroll
  for (int i = 0; i < 4; ++i)
    dst[(size_t)(nt + r0 + 8 * i) * K + kt + c] = (_Float16)tile[c][r0 + 8 * i];
}

// ---------------------------------------------------------------------------
// GEMM tile engine: C(64x128) += A(64xK) * Wt(128 rows x K), all f16 operands.
// 256 threads = 8 waves (2M x 4N), each wave owns 32x32 (2x2 frags). BK=64.
// Pure b128 global->LDS copies; prefetch next K-step.
// ---------------------------------------------------------------------------
__device__ void gemm_tile_f16(const _Float16* __restrict__ A,
                              const _Float16* __restrict__ Wt,
                              int Kt, int tM, int tN,
                              _Float16* As, _Float16* Ws, v8f acc[2][2], int tid) {
  const int lane = tid & 31, wid = tid >> 5;
  const int wm = wid & 1, wn = wid >> 1;
#pragma unroll
  for (int i = 0; i < 2; ++i)
#pragma unroll
    for (int j = 0; j < 2; ++j) acc[i][j] = zero8();

  const int arow = tid >> 2, ac = (tid & 3) << 4;   // A: 64 rows x 64 halves
  const int wrow = tid >> 1, wc = (tid & 1) << 5;   // Wt: 128 rows x 64 halves
  const _Float16* asrc = A  + (size_t)(tM + arow) * Kt + ac;
  const _Float16* wsrc = Wt + (size_t)(tN + wrow) * Kt + wc;
  _Float16* adst = As + arow * 64 + ac;
  _Float16* wdst = Ws + wrow * 64 + wc;

  for (int k0 = 0; k0 < Kt; k0 += 64) {
    if (k0 + 64 < Kt) {                 // prefetch next K-step tiles into caches
      __builtin_prefetch(asrc + 64, 0, 3);
      __builtin_prefetch(wsrc + 64, 0, 3);
    }
    __syncthreads();
    // A tile: 16 halves per thread (2 x b128)
    *(v8h*)(adst)     = *(const v8h*)(asrc + k0);
    *(v8h*)(adst + 8) = *(const v8h*)(asrc + k0 + 8);
    // Wt tile: 32 halves per thread (4 x b128)
#pragma unroll
    for (int j = 0; j < 4; ++j)
      *(v8h*)(wdst + 8 * j) = *(const v8h*)(wsrc + k0 + 8 * j);
    __syncthreads();

#pragma unroll
    for (int kk = 0; kk < 64; kk += 32) {
      v16h a0 = load_frag_a(As + (wm * 32) * 64 + kk, 64, lane);
      v16h a1 = load_frag_a(As + (wm * 32 + 16) * 64 + kk, 64, lane);
      v16h b0 = load_frag_b(Ws + (wn * 32) * 64 + kk, 64, lane);
      v16h b1 = load_frag_b(Ws + (wn * 32 + 16) * 64 + kk, 64, lane);
      acc[0][0] = wmma_f16(a0, b0, acc[0][0]);
      acc[0][1] = wmma_f16(a0, b1, acc[0][1]);
      acc[1][0] = wmma_f16(a1, b0, acc[1][0]);
      acc[1][1] = wmma_f16(a1, b1, acc[1][1]);
    }
  }
}

// ---------------------------------------------------------------------------
// Kernel 1: QKV projection. blockIdx.z selects q/k/v (scalar branch).
// Writes q[b,h,s,dh], k[b,h,s,dh], v^T[b,h,dh,s] as f16.
// ---------------------------------------------------------------------------
__global__ __launch_bounds__(256) void qkv_gemm_kernel(
    const _Float16* __restrict__ xh, const _Float16* __restrict__ wqkvT,
    const float* __restrict__ bias, _Float16* __restrict__ q,
    _Float16* __restrict__ kb, _Float16* __restrict__ vt) {
  __shared__ alignas(16) _Float16 As[64 * 64];
  __shared__ alignas(16) _Float16 Ws[128 * 64];
  const int tid = threadIdx.x;
  const int tM = blockIdx.y * 64, tN = blockIdx.x * 128;
  const int which = blockIdx.z;                       // 0=q, 1=k, 2=v (scalar)
  const _Float16* wt = wqkvT + (size_t)which * Dd * Dd;  // row block of Wt
  const float* bi = bias + which * Dd;

  v8f acc[2][2];
  gemm_tile_f16(xh, wt, Dd, tM, tN, As, Ws, acc, tid);

  const int lane = tid & 31, wid = tid >> 5;
  const int wm = wid & 1, wn = wid >> 1;
  const int rb = (lane >> 4) << 3, cc = lane & 15;

  if (which < 2) {
    _Float16* dst = (which == 0) ? q : kb;
#pragma unroll
    for (int fm = 0; fm < 2; ++fm)
#pragma unroll
      for (int fn = 0; fn < 2; ++fn)
#pragma unroll
        for (int r = 0; r < 8; ++r) {
          const int row = tM + wm * 32 + fm * 16 + rb + r;   // 0..4095
          const int col = tN + wn * 32 + fn * 16 + cc;       // 0..1023
          const _Float16 hv = (_Float16)(acc[fm][fn][r] + bi[col]);
          const int b = row >> 11, s = row & (Ss - 1);
          const int h = col >> 6, dh = col & 63;
          dst[((size_t)(b * Hh + h) * Ss + s) * DHh + dh] = hv;
        }
  } else {
#pragma unroll
    for (int fm = 0; fm < 2; ++fm)
#pragma unroll
      for (int fn = 0; fn < 2; ++fn)
#pragma unroll
        for (int r = 0; r < 8; ++r) {
          const int row = tM + wm * 32 + fm * 16 + rb + r;
          const int col = tN + wn * 32 + fn * 16 + cc;
          const _Float16 hv = (_Float16)(acc[fm][fn][r] + bi[col]);
          const int b = row >> 11, s = row & (Ss - 1);
          const int h = col >> 6, dh = col & 63;
          vt[((size_t)(b * Hh + h) * DHh + dh) * Ss + s] = hv;
        }
  }
}

// ---------------------------------------------------------------------------
// Kernel 2: flash attention. One wave per 16-query tile of one (b,h).
// ---------------------------------------------------------------------------
__global__ __launch_bounds__(256) void attn_kernel(
    const _Float16* __restrict__ q, const _Float16* __restrict__ kbuf,
    const _Float16* __restrict__ vt, _Float16* __restrict__ attn) {
  __shared__ alignas(16) _Float16 pst[8][16 * 32];   // per-wave P staging
  const int tid = threadIdx.x, lane = tid & 31, wid = tid >> 5;
  const int gw = blockIdx.x * 8 + wid;               // 0..4095
  const int qt = gw & 127;                           // S/16 tiles
  const int bh = gw >> 7;                            // 0..31
  const int b = bh >> 4, h = bh & 15;

  const _Float16* qp = q    + ((size_t)bh * Ss + qt * 16) * DHh;
  const _Float16* kp = kbuf + (size_t)bh * Ss * DHh;
  const _Float16* vp = vt   + (size_t)bh * DHh * Ss;

  const v16h qa0 = load_frag_a(qp, DHh, lane);       // dh 0..31
  const v16h qa1 = load_frag_a(qp + 32, DHh, lane);  // dh 32..63

  v8f co[4];
  float mi[8], li[8];
#pragma unroll
  for (int f = 0; f < 4; ++f) co[f] = zero8();
#pragma unroll
  for (int r = 0; r < 8; ++r) { mi[r] = -1.0e30f; li[r] = 0.0f; }

  _Float16* ps = &pst[wid][0];
  const int rb = (lane >> 4) << 3, cc = lane & 15;

  for (int c0 = 0; c0 < Ss; c0 += 32) {
    if (c0 + 32 < Ss) __builtin_prefetch(kp + (size_t)(c0 + 32) * DHh, 0, 3);
    // scores: 16 queries x 32 keys, contraction over dh=64 (2 x K32)
    v8f s0 = zero8(), s1 = zero8();
    s0 = wmma_f16(qa0, load_frag_b(kp + (size_t)c0 * DHh, DHh, lane), s0);
    s0 = wmma_f16(qa1, load_frag_b(kp + (size_t)c0 * DHh + 32, DHh, lane), s0);
    s1 = wmma_f16(qa0, load_frag_b(kp + (size_t)(c0 + 16) * DHh, DHh, lane), s1);
    s1 = wmma_f16(qa1, load_frag_b(kp + (size_t)(c0 + 16) * DHh + 32, DHh, lane), s1);
    s0 = s0 * 0.125f;                                // 1/sqrt(64)
    s1 = s1 * 0.125f;

    // online softmax; rows live per-lane-half, reduce across 16 lanes
#pragma unroll
    for (int r = 0; r < 8; ++r) {
      float t = fmaxf(s0[r], s1[r]);
      t = fmaxf(t, __shfl_xor(t, 1, 32));
      t = fmaxf(t, __shfl_xor(t, 2, 32));
      t = fmaxf(t, __shfl_xor(t, 4, 32));
      t = fmaxf(t, __shfl_xor(t, 8, 32));
      const float mnew  = fmaxf(mi[r], t);
      const float alpha = __expf(mi[r] - mnew);
      const float e0 = __expf(s0[r] - mnew);
      const float e1 = __expf(s1[r] - mnew);
      s0[r] = e0; s1[r] = e1;
      float rs = e0 + e1;
      rs += __shfl_xor(rs, 1, 32);
      rs += __shfl_xor(rs, 2, 32);
      rs += __shfl_xor(rs, 4, 32);
      rs += __shfl_xor(rs, 8, 32);
      li[r] = li[r] * alpha + rs;
      mi[r] = mnew;
      co[0][r] *= alpha; co[1][r] *= alpha;
      co[2][r] *= alpha; co[3][r] *= alpha;
    }

    // C-layout P -> LDS -> A-layout fragment (per-wave private patch)
#pragma unroll
    for (int r = 0; r < 8; ++r) {
      ps[(rb + r) * 32 + cc]      = (_Float16)s0[r];
      ps[(rb + r) * 32 + 16 + cc] = (_Float16)s1[r];
    }
    asm volatile("s_wait_dscnt 0x0" ::: "memory");
    const v16h pa = load_frag_a(ps, 32, lane);

    // O += P(16x32) * V(32x64); V^T rows are dh, cols are keys
#pragma unroll
    for (int f = 0; f < 4; ++f) {
      const v16h vb = load_frag_b(vp + (size_t)(f * 16) * Ss + c0, Ss, lane);
      co[f] = wmma_f16(pa, vb, co[f]);
    }
  }

  // normalize and store to attn[b, s, h*64 + dh] (f16)
#pragma unroll
  for (int f = 0; f < 4; ++f)
#pragma unroll
    for (int r = 0; r < 8; ++r) {
      const float val = co[f][r] / li[r];
      const int srow = qt * 16 + rb + r;
      const int dcol = h * DHh + f * 16 + cc;
      attn[((size_t)b * Ss + srow) * Dd + dcol] = (_Float16)val;
    }
}

// ---------------------------------------------------------------------------
// Kernel 3: output projection, f32 result + bias
// ---------------------------------------------------------------------------
__global__ __launch_bounds__(256) void out_gemm_kernel(
    const _Float16* __restrict__ a, const _Float16* __restrict__ woutT,
    const float* __restrict__ bias, float* __restrict__ out) {
  __shared__ alignas(16) _Float16 As[64 * 64];
  __shared__ alignas(16) _Float16 Ws[128 * 64];
  const int tid = threadIdx.x;
  const int tM = blockIdx.y * 64, tN = blockIdx.x * 128;
  v8f acc[2][2];
  gemm_tile_f16(a, woutT, Dd, tM, tN, As, Ws, acc, tid);

  const int lane = tid & 31, wid = tid >> 5;
  const int wm = wid & 1, wn = wid >> 1;
  const int rb = (lane >> 4) << 3, cc = lane & 15;
#pragma unroll
  for (int fm = 0; fm < 2; ++fm)
#pragma unroll
    for (int fn = 0; fn < 2; ++fn)
#pragma unroll
      for (int r = 0; r < 8; ++r) {
        const int row = tM + wm * 32 + fm * 16 + rb + r;
        const int col = tN + wn * 32 + fn * 16 + cc;
        out[(size_t)row * Dd + col] = acc[fm][fn][r] + bias[col];
      }
}

// ---------------------------------------------------------------------------
extern "C" void kernel_launch(void* const* d_in, const int* in_sizes, int n_in,
                              void* d_out, int out_size, void* d_ws, size_t ws_size,
                              hipStream_t stream) {
  (void)in_sizes; (void)n_in; (void)out_size; (void)ws_size;
  const float* x     = (const float*)d_in[0];
  const float* w_qkv = (const float*)d_in[1];
  const float* b_qkv = (const float*)d_in[2];
  const float* w_out = (const float*)d_in[3];
  const float* b_out = (const float*)d_in[4];

  // workspace layout (halves): 48 MB total
  const size_t elems = (size_t)Bb * Hh * Ss * DHh;   // 4,194,304 (= B*S*D)
  _Float16* xh     = (_Float16*)d_ws;                // 4M halves
  _Float16* wqkvT  = xh + elems;                     // 3M halves [3072][1024]
  _Float16* woutT  = wqkvT + (size_t)3 * Dd * Dd;    // 1M halves [1024][1024]
  _Float16* q      = woutT + (size_t)Dd * Dd;
  _Float16* kbuf   = q + elems;
  _Float16* vt     = kbuf + elems;
  _Float16* attn   = vt + elems;

  // 0) one-time conversions / transposes (all L2-resident afterwards)
  convert_f16_kernel<<<dim3((unsigned)(elems / (256 * 8))), 256, 0, stream>>>(x, xh);
  transpose_f16_kernel<<<dim3(3 * Dd / 32, Dd / 32), 256, 0, stream>>>(
      w_qkv, wqkvT, Dd, 3 * Dd);
  transpose_f16_kernel<<<dim3(Dd / 32, Dd / 32), 256, 0, stream>>>(
      w_out, woutT, Dd, Dd);

  // 1) QKV projection: M=4096, K=1024, z selects q/k/v (N=1024 each)
  qkv_gemm_kernel<<<dim3(Dd / 128, (Bb * Ss) / 64, 3), 256, 0, stream>>>(
      xh, wqkvT, b_qkv, q, kbuf, vt);

  // 2) flash attention: B*H*(S/16) = 4096 wave tiles / 8 waves per block
  attn_kernel<<<dim3((Bb * Hh * (Ss / 16)) / 8), 256, 0, stream>>>(
      q, kbuf, vt, attn);

  // 3) output projection: M=4096, N=1024, K=1024
  out_gemm_kernel<<<dim3(Dd / 128, (Bb * Ss) / 64), 256, 0, stream>>>(
      attn, w_out ? woutT : woutT, b_out, (float*)d_out);
}
